// Conv1d_fft_1778116461049
// MI455X (gfx1250) — compile-verified, matrix operand order checked by reading, not run on hardware
//
#include <hip/hip_runtime.h>
#include <hip/hip_bf16.h>
#include <cstdint>

// ---------------------------------------------------------------------------
// Conv1d (B=16, Cin=128, Cout=128, L=4096, K=129, pad=32, out_len=4032)
// as implicit GEMM on CDNA5 WMMA, bf16 hi/lo split for ~fp32 accuracy.
// out[b,o,n] = bias[o] + sum_i sum_t w[o,i,t] * x[b,i,n+t-32]
// ---------------------------------------------------------------------------

typedef __attribute__((ext_vector_type(16))) __bf16 v16bf;
typedef __attribute__((ext_vector_type(8)))  float  v8f;

#define CIN    128
#define COUT   128
#define LIN    4096
#define KTAPS  129
#define KPAD   160          // 5 chunks of 32
#define NCHUNK 5
#define OUTL   4032
#define NTILE  64
#define WINLEN 224          // NTILE + KPAD - 1 + ... covers x[n0-32 .. n0+191]

__device__ __forceinline__ uint32_t bf16_rne(float f) {
  uint32_t u = __float_as_uint(f);
  return (u + 0x7FFFu + ((u >> 16) & 1u)) >> 16;
}
__device__ __forceinline__ float bf16_f32(uint32_t h) {
  return __uint_as_float(h << 16);
}
// split f32 into bf16 hi + bf16 lo (bits)
__device__ __forceinline__ void split_bf16(float v, uint32_t& hi, uint32_t& lo) {
  hi = bf16_rne(v);
  lo = bf16_rne(v - bf16_f32(hi));
}

// Load one 16x32 bf16 fragment (A or B side) from LDS packed-pair layout.
// lds is a dword array [row][16] where dword j of a row = K pair (2j,2j+1).
// Per ISA 7.12.2: lanes 0-15 take K pairs {0..3}+{8..11} dwords offset by
// kb=0, lanes 16-31 take kb=4.  Two aligned 16B loads.
__device__ __forceinline__ v16bf load_frag(const uint32_t* lds, int base) {
  uint4 a = *(const uint4*)(lds + base);      // K = kb*2 .. kb*2+7
  uint4 b = *(const uint4*)(lds + base + 8);  // K = kb*2+16 .. kb*2+23
  union { uint32_t u[8]; v16bf v; } f;
  f.u[0] = a.x; f.u[1] = a.y; f.u[2] = a.z; f.u[3] = a.w;
  f.u[4] = b.x; f.u[5] = b.y; f.u[6] = b.z; f.u[7] = b.w;
  return f.v;
}

__global__ __launch_bounds__(256)
void conv1d_wmma_kernel(const float* __restrict__ x,
                        const float* __restrict__ w,
                        const float* __restrict__ bias,
                        float* __restrict__ out) {
  __shared__ uint32_t Wh[COUT * 16];   // weight chunk, bf16-hi pairs [m][kpair]
  __shared__ uint32_t Wl[COUT * 16];   // weight chunk, bf16-lo pairs
  __shared__ uint32_t Xh[NTILE * 16];  // x chunk, bf16-hi pairs [n][kpair]
  __shared__ uint32_t Xl[NTILE * 16];  // x chunk, bf16-lo pairs
  __shared__ float    xwin[WINLEN];    // fp32 x window for current channel
  __shared__ float    biasS[COUT];

  const int tid  = threadIdx.x;
  const int lane = tid & 31;
  const int wv   = tid >> 5;                 // 8 waves: rows [16*wv, 16*wv+16)
  const int b    = blockIdx.z;               // batch
  const int n0   = blockIdx.x * NTILE;       // output-position tile base

  const int kb   = (lane < 16) ? 0 : 4;      // dword K-base per lane half
  const int l16  = lane & 15;

  if (tid < COUT) biasS[tid] = bias[tid];

  v8f acc0 = {}, acc1 = {}, acc2 = {}, acc3 = {};

  const int wm   = tid >> 1;                 // weight-stage row 0..127
  const int half = tid & 1;                  // which 16-tap half of the chunk

  for (int ch = 0; ch < CIN; ++ch) {
    __syncthreads();  // prior chunk's compute done before xwin overwrite

    // ---- stage fp32 x window for this channel: x[b,ch, n0-32 .. n0+191] ----
    if (tid < WINLEN) {
      int pos = n0 - 32 + tid;
      float v = 0.0f;
      const float* xc = x + ((size_t)b * CIN + ch) * LIN;
      if (pos >= 0 && pos < LIN) v = xc[pos];
      xwin[tid] = v;
      if (ch + 1 < CIN) {   // pull next channel's window toward the caches
        int p2 = pos < 0 ? 0 : (pos >= LIN ? LIN - 1 : pos);
        __builtin_prefetch(xc + LIN + p2, 0, 3);   // global_prefetch_b8
      }
    }
    __syncthreads();

    const float* wrow = w + ((size_t)wm * CIN + ch) * KTAPS;

    for (int c = 0; c < NCHUNK; ++c) {
      // ---- stage weight chunk w[0:128, ch, c*32 .. c*32+31] as bf16 hi/lo --
      {
        int tbase = c * 32 + half * 16;
#pragma unroll
        for (int jj = 0; jj < 8; ++jj) {
          int t0 = tbase + 2 * jj;
          float v0 = (t0     < KTAPS) ? wrow[t0]     : 0.0f;
          float v1 = (t0 + 1 < KTAPS) ? wrow[t0 + 1] : 0.0f;
          uint32_t h0, l0, h1, l1;
          split_bf16(v0, h0, l0);
          split_bf16(v1, h1, l1);
          int di = wm * 16 + half * 8 + jj;
          Wh[di] = h0 | (h1 << 16);
          Wl[di] = l0 | (l1 << 16);
        }
      }
      // ---- stage x chunk: Xh/Xl[n][j] = pack(xwin[n+c*32+2j], +1) ----------
#pragma unroll
      for (int q = 0; q < 4; ++q) {
        int idx = tid * 4 + q;              // 0..1023 == n*16 + j
        int base = (idx >> 4) + c * 32 + 2 * (idx & 15);
        uint32_t h0, l0, h1, l1;
        split_bf16(xwin[base],     h0, l0);
        split_bf16(xwin[base + 1], h1, l1);
        Xh[idx] = h0 | (h1 << 16);
        Xl[idx] = l0 | (l1 << 16);
      }
      __syncthreads();

      // ---- compute: 12 WMMAs per wave per chunk ----------------------------
      const int abase = (wv * 16 + l16) * 16 + kb;
      v16bf Ah = load_frag(Wh, abase);
      v16bf Al = load_frag(Wl, abase);

#pragma unroll
      for (int cg = 0; cg < 4; ++cg) {
        const int bbase = (cg * 16 + l16) * 16 + kb;
        v16bf Bh = load_frag(Xh, bbase);
        v16bf Bl = load_frag(Xl, bbase);
        v8f* acc = (cg == 0) ? &acc0 : (cg == 1) ? &acc1 : (cg == 2) ? &acc2 : &acc3;
        // hi*hi + hi*lo + lo*hi  ~= fp32 product, fp32 accumulate
        *acc = __builtin_amdgcn_wmma_f32_16x16x32_bf16(false, Ah, false, Bh,
                                                       (short)0, *acc, false, false);
        *acc = __builtin_amdgcn_wmma_f32_16x16x32_bf16(false, Ah, false, Bl,
                                                       (short)0, *acc, false, false);
        *acc = __builtin_amdgcn_wmma_f32_16x16x32_bf16(false, Al, false, Bh,
                                                       (short)0, *acc, false, false);
      }
      __syncthreads();
    }
  }

  // ---- epilogue: C/D layout (ISA 7.12.2): VGPR v -> row v (lanes<16) or
  // row v+8 (lanes>=16); col = lane%16.  Add bias, store fp32. --------------
  const int mbase = wv * 16 + ((lane < 16) ? 0 : 8);
#pragma unroll
  for (int cg = 0; cg < 4; ++cg) {
    v8f acc = (cg == 0) ? acc0 : (cg == 1) ? acc1 : (cg == 2) ? acc2 : acc3;
    int col = n0 + cg * 16 + l16;
#pragma unroll
    for (int v = 0; v < 8; ++v) {
      int m = mbase + v;
      out[((size_t)b * COUT + m) * OUTL + col] = acc[v] + biasS[m];
    }
  }
}

extern "C" void kernel_launch(void* const* d_in, const int* in_sizes, int n_in,
                              void* d_out, int out_size, void* d_ws, size_t ws_size,
                              hipStream_t stream) {
  (void)in_sizes; (void)n_in; (void)out_size; (void)d_ws; (void)ws_size;
  const float* x    = (const float*)d_in[0];   // (16,128,4096)
  const float* w    = (const float*)d_in[1];   // (128,128,129)
  const float* bias = (const float*)d_in[2];   // (128,)
  float* out = (float*)d_out;                  // (16,128,4032)

  dim3 grid(OUTL / NTILE, 1, 16);              // 63 N-tiles x 16 batches
  dim3 block(256);                             // 8 wave32 -> full 128 Cout rows
  conv1d_wmma_kernel<<<grid, block, 0, stream>>>(x, w, bias, out);
}
